// Attention_9242769622327
// MI455X (gfx1250) — matrix operand order repaired and checked by reading, not compiled, not run on hardware
//
#include <hip/hip_runtime.h>
#include <cstdint>
#include <cstddef>

// ---------------------------------------------------------------------------
// MI455X (gfx1250) implementation.
// Wave32, WMMA bf16 16x16x32 with f32 accumulation, async global->LDS staging.
// ---------------------------------------------------------------------------

typedef __bf16 bf16_t;
typedef __attribute__((ext_vector_type(16))) __bf16        v16bf;
typedef __attribute__((ext_vector_type(8)))  float         v8f;
typedef __attribute__((ext_vector_type(4)))  unsigned int  u32x4;

union FragAB { v16bf v; u32x4 u[2]; };

static constexpr int kB   = 4;
static constexpr int kS   = 8192;
static constexpr int kE   = 1024;
static constexpr int kH   = 16;
static constexpr int kEh  = 64;          // head dim
static constexpr int kM   = kB * kS;     // 32768 rows
static constexpr int kN1  = 3 * kE;      // 3072 qkv cols
static constexpr int kP   = kB * kH;     // 64 (b,h) pairs
static constexpr int kSplit = 8;         // S-splits for score GEMM

static __device__ __forceinline__ unsigned int bf16_bits(float f) {
  bf16_t b = (bf16_t)f;
  return (unsigned int)__builtin_bit_cast(unsigned short, b);
}

// Async copy of 16 bytes global -> LDS (ASYNCcnt-tracked, per-lane addresses).
// LDS operand is the byte offset = low 32 bits of the generic LDS pointer.
static __device__ __forceinline__ void async_copy_b128(const bf16_t* gsrc,
                                                       bf16_t* ldst) {
  unsigned int lds_off = (unsigned int)(size_t)ldst;
  unsigned long long gaddr = (unsigned long long)(size_t)gsrc;
  asm volatile("global_load_async_to_lds_b128 %0, %1, off"
               :
               : "v"(lds_off), "v"(gaddr)
               : "memory");
}

static __device__ __forceinline__ void wait_async0() {
  asm volatile("s_wait_asynccnt 0" ::: "memory");
}

// ---------------------------------------------------------------------------
// fp32 -> bf16 conversion
// ---------------------------------------------------------------------------
__global__ __launch_bounds__(256) void f32_to_bf16(
    const float* __restrict__ src, bf16_t* __restrict__ dst, size_t n) {
  size_t i = (size_t)blockIdx.x * 256 + threadIdx.x;
  if (i < n) dst[i] = (bf16_t)src[i];
}

// ---------------------------------------------------------------------------
// 128x128-tile bf16 GEMM:  C[M,N] = A[M,K] * Bw[N,K]^T
// A and Bw are both K-contiguous, matching WMMA A-fragment layout (lane = row,
// K striped through VGPRs) and B-fragment layout (lane = column of B == row of
// Bw). 256 threads = 8 waves; wave tile 32(M) x 64(N) => 2x4 C fragments.
// Double-buffered LDS, async global->LDS copies overlapped with WMMA.
// MODE 0: plain f32 store to C (ld = N).
// MODE 1: QKV epilogue: n<1024 -> q (f32, [m,1024]);
//         1024..2047  -> kT bf16, scaled by 1/4, layout [bh][64][S];
//         2048..3071  -> vT bf16, layout [bh][64][S].
// ---------------------------------------------------------------------------
template <int MODE>
__global__ __launch_bounds__(256) void gemm_bf16_128x128(
    const bf16_t* __restrict__ A, const bf16_t* __restrict__ Bw,
    float* __restrict__ C, bf16_t* __restrict__ kT, bf16_t* __restrict__ vT,
    int K, int N) {
  constexpr int KC  = 64;   // K elements per stage
  constexpr int LDT = 72;   // row pitch in elements (144B, 16B aligned)
  __shared__ __align__(16) bf16_t As[2][128 * LDT];
  __shared__ __align__(16) bf16_t Bs[2][128 * LDT];

  const int tid  = threadIdx.x;
  const int lane = tid & 31;
  const int wave = tid >> 5;
  const int l15  = lane & 15;
  const int lhi  = lane >> 4;

  const int bm  = blockIdx.x * 128;
  const int bn  = blockIdx.y * 128;
  const int wm0 = (wave & 3) * 32;
  const int wn0 = (wave >> 2) * 64;

  // tile-loader assignment: 2 threads per row, each moves 32 elems (4x b128)
  const int lr = tid >> 1;
  const int lc = (tid & 1) * 32;
  const bf16_t* gA = A  + (size_t)(bm + lr) * K + lc;
  const bf16_t* gB = Bw + (size_t)(bn + lr) * K + lc;

  const v8f vzero = {0.f, 0.f, 0.f, 0.f, 0.f, 0.f, 0.f, 0.f};
  v8f acc[2][4];
#pragma unroll
  for (int i = 0; i < 2; ++i)
#pragma unroll
    for (int j = 0; j < 4; ++j) acc[i][j] = vzero;

  const int nStages = K / KC;

  // prologue: stage 0
#pragma unroll
  for (int c = 0; c < 4; ++c) {
    async_copy_b128(gA + c * 8, &As[0][lr * LDT + lc + c * 8]);
    async_copy_b128(gB + c * 8, &Bs[0][lr * LDT + lc + c * 8]);
  }
  wait_async0();
  __syncthreads();

  for (int st = 0; st < nStages; ++st) {
    const int cur = st & 1;
    if (st + 1 < nStages) {
      const int k1 = (st + 1) * KC;
#pragma unroll
      for (int c = 0; c < 4; ++c) {
        async_copy_b128(gA + k1 + c * 8, &As[cur ^ 1][lr * LDT + lc + c * 8]);
        async_copy_b128(gB + k1 + c * 8, &Bs[cur ^ 1][lr * LDT + lc + c * 8]);
      }
      if (st + 2 < nStages) {
        __builtin_prefetch(gA + (st + 2) * KC);  // global_prefetch_b8
        __builtin_prefetch(gB + (st + 2) * KC);
      }
    }

#pragma unroll
    for (int ks = 0; ks < KC; ks += 32) {
      FragAB a[2], b[4];
#pragma unroll
      for (int fm = 0; fm < 2; ++fm) {
        const int ar = wm0 + fm * 16 + l15;
        const int ak = ks + lhi * 8;  // A: lanes>=16 hold K=8..15 / 24..31
        a[fm].u[0] = *(const u32x4*)(&As[cur][ar * LDT + ak]);
        a[fm].u[1] = *(const u32x4*)(&As[cur][ar * LDT + ak + 16]);
      }
#pragma unroll
      for (int fn = 0; fn < 4; ++fn) {
        const int br = wn0 + fn * 16 + l15;
        const int bk = ks + lhi * 8;  // B: lanes>=16 hold K=8..15 / 24..31
        b[fn].u[0] = *(const u32x4*)(&Bs[cur][br * LDT + bk]);
        b[fn].u[1] = *(const u32x4*)(&Bs[cur][br * LDT + bk + 16]);
      }
#pragma unroll
      for (int fm = 0; fm < 2; ++fm)
#pragma unroll
        for (int fn = 0; fn < 4; ++fn)
          acc[fm][fn] = __builtin_amdgcn_wmma_f32_16x16x32_bf16(
              false, a[fm].v, false, b[fn].v, (short)0, acc[fm][fn],
              false, false);
    }

    if (st + 1 < nStages) {
      wait_async0();    // my next-stage async copies have landed in LDS
      __syncthreads();  // everyone's have; prev buffer free for reuse
    }
  }

  // epilogue: C fragment layout: lanes 0-15 -> M = g, lanes 16-31 -> M = g+8;
  // N = lane & 15.
#pragma unroll
  for (int fm = 0; fm < 2; ++fm) {
#pragma unroll
    for (int fn = 0; fn < 4; ++fn) {
      const int mb = bm + wm0 + fm * 16 + lhi * 8;
      const int n  = bn + wn0 + fn * 16 + l15;
#pragma unroll
      for (int g = 0; g < 8; ++g) {
        const int m = mb + g;
        const float val = acc[fm][fn][g];
        if constexpr (MODE == 0) {
          C[(size_t)m * N + n] = val;
        } else {
          if (n < kE) {
            C[(size_t)m * kE + n] = val;  // q, fp32, [S_total, E]
          } else {
            const int bb = m >> 13;          // batch
            const int ss = m & (kS - 1);     // seq position
            if (n < 2 * kE) {
              const int kf = n - kE;
              const int p  = (bb << 4) | (kf >> 6);
              kT[(size_t)((p << 6) | (kf & 63)) * kS + ss] =
                  (bf16_t)(val * 0.25f);     // fold k/4 here
            } else {
              const int cf = n - 2 * kE;
              const int p  = (bb << 4) | (cf >> 6);
              vT[(size_t)((p << 6) | (cf & 63)) * kS + ss] = (bf16_t)val;
            }
          }
        }
      }
    }
  }
}

// ---------------------------------------------------------------------------
// score_pre[p][kk][cc] = sum_s kT[p][kk][s] * vT[p][cc][s]   (64x64 per pair)
// Split-S over kSplit blocks -> partials in spart. 4 waves, wave owns 16 rows.
// kT/vT are reduction-contiguous, so A and B fragments are straight 16B loads.
// ---------------------------------------------------------------------------
__global__ __launch_bounds__(128) void score_gemm(
    const bf16_t* __restrict__ kT, const bf16_t* __restrict__ vT,
    float* __restrict__ spart) {
  const int p    = blockIdx.x;
  const int j    = blockIdx.y;
  const int lane = threadIdx.x & 31;
  const int wave = threadIdx.x >> 5;
  const int l15  = lane & 15;
  const int lhi  = lane >> 4;

  const v8f vzero = {0.f, 0.f, 0.f, 0.f, 0.f, 0.f, 0.f, 0.f};
  v8f acc[4];
#pragma unroll
  for (int fn = 0; fn < 4; ++fn) acc[fn] = vzero;

  const bf16_t* ka = kT + (size_t)(p * 64 + wave * 16 + l15) * kS + lhi * 8;
  const int sBeg = j * (kS / kSplit);
  const int sEnd = sBeg + (kS / kSplit);
  for (int s0 = sBeg; s0 < sEnd; s0 += 32) {
    FragAB a;
    a.u[0] = *(const u32x4*)(ka + s0);
    a.u[1] = *(const u32x4*)(ka + s0 + 16);
#pragma unroll
    for (int fn = 0; fn < 4; ++fn) {
      FragAB b;
      const bf16_t* vb =
          vT + (size_t)(p * 64 + fn * 16 + l15) * kS + lhi * 8 + s0;
      b.u[0] = *(const u32x4*)(vb);
      b.u[1] = *(const u32x4*)(vb + 16);
      acc[fn] = __builtin_amdgcn_wmma_f32_16x16x32_bf16(
          false, a.v, false, b.v, (short)0, acc[fn], false, false);
    }
  }
#pragma unroll
  for (int fn = 0; fn < 4; ++fn)
#pragma unroll
    for (int g = 0; g < 8; ++g) {
      const int m = wave * 16 + lhi * 8 + g;
      const int n = fn * 16 + l15;
      spart[((size_t)(p * kSplit + j) * 64 + m) * 64 + n] = acc[fn][g];
    }
}

// ---------------------------------------------------------------------------
// Reduce split partials, softmax each 64-row over c, emit row-sum of softmax
// (the reference's einsum over c collapses to exactly this).
// ---------------------------------------------------------------------------
__global__ __launch_bounds__(64) void score_softmax_rowsum(
    const float* __restrict__ spart, float* __restrict__ rbuf) {
  const int p  = blockIdx.x;
  const int kk = threadIdx.x;
  float v[64];
#pragma unroll
  for (int c = 0; c < 64; ++c) {
    float s = 0.f;
#pragma unroll
    for (int j = 0; j < kSplit; ++j)
      s += spart[((size_t)(p * kSplit + j) * 64 + kk) * 64 + c];
    v[c] = s;
  }
  float mx = -__builtin_inff();
#pragma unroll
  for (int c = 0; c < 64; ++c) mx = fmaxf(mx, v[c]);
  float d = 0.f;
#pragma unroll
  for (int c = 0; c < 64; ++c) { v[c] = __expf(v[c] - mx); d += v[c]; }
  const float inv = 1.f / d;
  float r = 0.f;
#pragma unroll
  for (int c = 0; c < 64; ++c) r += v[c] * inv;
  rbuf[p * 64 + kk] = r;
}

// ---------------------------------------------------------------------------
// attn[sg, h*64+kk] = softmax_kk( q[sg, h*64+kk] * 0.25 * r[b*16+h, kk] )
// One thread per (row, head); output packed bf16 (GEMM-2 A operand).
// ---------------------------------------------------------------------------
__global__ __launch_bounds__(256) void attn_softmax(
    const float* __restrict__ qbuf, const float* __restrict__ rbuf,
    bf16_t* __restrict__ abuf) {
  const size_t idx = (size_t)blockIdx.x * 256 + threadIdx.x;
  const int sg = (int)(idx >> 4);
  const int h  = (int)(idx & 15);
  const int bb = sg >> 13;
  const float* q = qbuf + (size_t)sg * kE + h * kEh;
  const float* r = rbuf + (size_t)((bb << 4) | h) * kEh;

  float v[kEh];
  float mx = -__builtin_inff();
#pragma unroll
  for (int i = 0; i < kEh; ++i) {
    const float t = q[i] * 0.25f * r[i];
    v[i] = t;
    mx = fmaxf(mx, t);
  }
  float d = 0.f;
#pragma unroll
  for (int i = 0; i < kEh; ++i) { v[i] = __expf(v[i] - mx); d += v[i]; }
  const float inv = 1.f / d;

  unsigned int* dst = (unsigned int*)(abuf + (size_t)sg * kE + h * kEh);
#pragma unroll
  for (int i = 0; i < kEh / 2; ++i) {
    const unsigned int lo = bf16_bits(v[2 * i] * inv);
    const unsigned int hi = bf16_bits(v[2 * i + 1] * inv);
    dst[i] = (hi << 16) | lo;
  }
}

// ---------------------------------------------------------------------------
// Host-side orchestration
// ---------------------------------------------------------------------------
extern "C" void kernel_launch(void* const* d_in, const int* in_sizes, int n_in,
                              void* d_out, int out_size, void* d_ws,
                              size_t ws_size, hipStream_t stream) {
  (void)in_sizes; (void)n_in; (void)out_size; (void)ws_size;
  const float* x  = (const float*)d_in[0];
  const float* W1 = (const float*)d_in[1];
  const float* W2 = (const float*)d_in[2];
  // d_in[3] = heads (==16), baked into constants.
  float* out = (float*)d_out;

  char* ws = (char*)d_ws;
  size_t off = 0;
  auto carve = [&](size_t bytes) -> void* {
    off = (off + 255) & ~(size_t)255;
    void* p = ws + off;
    off += bytes;
    return p;
  };

  bf16_t* xb  = (bf16_t*)carve((size_t)kM * kE * 2);            // 64 MiB
  bf16_t* w1b = (bf16_t*)carve((size_t)kN1 * kE * 2);           // 6 MiB
  bf16_t* w2b = (bf16_t*)carve((size_t)kE * kE * 2);            // 2 MiB
  float*  qb  = (float*) carve((size_t)kM * kE * 4);            // 128 MiB
  bf16_t* kT  = (bf16_t*)carve((size_t)kP * kEh * kS * 2);      // 64 MiB
  bf16_t* vT  = (bf16_t*)carve((size_t)kP * kEh * kS * 2);      // 64 MiB
  float*  sp  = (float*) carve((size_t)kP * kSplit * kEh * kEh * 4);
  float*  rb  = (float*) carve((size_t)kP * kEh * 4);
  bf16_t* ab  = xb;  // reuse x's bf16 buffer: x is dead after GEMM-1

  const size_t nx  = (size_t)kM * kE;
  const size_t nw1 = (size_t)kN1 * kE;
  const size_t nw2 = (size_t)kE * kE;
  f32_to_bf16<<<(unsigned)((nx  + 255) / 256), 256, 0, stream>>>(x,  xb,  nx);
  f32_to_bf16<<<(unsigned)((nw1 + 255) / 256), 256, 0, stream>>>(W1, w1b, nw1);
  f32_to_bf16<<<(unsigned)((nw2 + 255) / 256), 256, 0, stream>>>(W2, w2b, nw2);

  // qkv projection (emits q fp32, kT/4 + vT bf16 transposed)
  gemm_bf16_128x128<1><<<dim3(kM / 128, kN1 / 128), 256, 0, stream>>>(
      xb, w1b, qb, kT, vT, kE, kN1);

  // 64x64 score matrices per (b,h), split over S
  score_gemm<<<dim3(kP, kSplit), 128, 0, stream>>>(kT, vT, sp);
  score_softmax_rowsum<<<kP, kEh, 0, stream>>>(sp, rb);

  // attn = softmax(q/4 * rowsum), packed bf16
  attn_softmax<<<(kM * kH) / 256, 256, 0, stream>>>(qb, rb, ab);

  // output projection
  gemm_bf16_128x128<0><<<dim3(kM / 128, kE / 128), 256, 0, stream>>>(
      ab, w2b, out, nullptr, nullptr, kE, kE);
}
